// Attention_2920577761563
// MI455X (gfx1250) — matrix operand order repaired and checked by reading, not compiled, not run on hardware
//
#include <hip/hip_runtime.h>
#include <hip/hip_bf16.h>
#include <math.h>

typedef __bf16 bf16;
typedef __attribute__((ext_vector_type(16))) __bf16 bf16x16;
typedef __attribute__((ext_vector_type(8)))  __bf16 bf16x8;
typedef __attribute__((ext_vector_type(8)))  float  v8f;

union AFrag {
    bf16x16 v;
    bf16x8  h[2];
};

// ---------------------------------------------------------------------------
// fp32 -> bf16 conversion
// ---------------------------------------------------------------------------
__global__ __launch_bounds__(256) void cvt_f32_bf16(const float* __restrict__ in,
                                                    bf16* __restrict__ out, long n) {
    long i = (long)blockIdx.x * blockDim.x + threadIdx.x;
    if (i < n) out[i] = (bf16)in[i];
}

// ---------------------------------------------------------------------------
// GEMM: out[M,N] = A[M,K](bf16, row-major) @ W[N,K](bf16, row-major)^T + bias
// One wave computes a 64x64 tile via 4x4 grid of 16x16 WMMA accumulators.
// K contiguous in both operands -> fragment loads are plain vector loads.
// ---------------------------------------------------------------------------
__global__ __launch_bounds__(256)
void gemm_bf16_wmma(const bf16* __restrict__ A, const bf16* __restrict__ W,
                    const float* __restrict__ bias, float* __restrict__ out,
                    int M, int N, int K) {
    const int lane = threadIdx.x & 31;
    const int wave = threadIdx.x >> 5;
    const int col  = lane & 15;
    const int half = lane >> 4;

    const int tilesN = N >> 6;
    int wg = blockIdx.x * 8 + wave;
    int tn = (wg % tilesN) * 64;
    int tm = (wg / tilesN) * 64;

    v8f acc[4][4] = {};

    for (int k0 = 0; k0 < K; k0 += 32) {
        if (k0 + 32 < K) {
            __builtin_prefetch(A + (size_t)(tm + col) * K + k0 + 32, 0, 1);
            __builtin_prefetch(W + (size_t)(tn + col) * K + k0 + 32, 0, 1);
        }
        AFrag a[4];
#pragma unroll
        for (int i = 0; i < 4; ++i) {
            const bf16* p = A + (size_t)(tm + 16 * i + col) * K + k0 + 8 * half;
            a[i].h[0] = *(const bf16x8*)p;
            a[i].h[1] = *(const bf16x8*)(p + 16);
        }
#pragma unroll
        for (int j = 0; j < 4; ++j) {
            AFrag b;
            b.h[0] = *(const bf16x8*)(W + (size_t)(tn + 16 * j + col) * K + k0 + 8 * half);
            b.h[1] = *(const bf16x8*)(W + (size_t)(tn + 16 * j + col) * K + k0 + 8 * half + 16);
#pragma unroll
            for (int i = 0; i < 4; ++i) {
                acc[i][j] = __builtin_amdgcn_wmma_f32_16x16x32_bf16(
                    false, a[i].v, false, b.v, (short)0, acc[i][j], false, false);
            }
        }
    }

#pragma unroll
    for (int j = 0; j < 4; ++j) {
        int c = tn + 16 * j + col;
        float bs = bias[c];
#pragma unroll
        for (int i = 0; i < 4; ++i) {
#pragma unroll
            for (int r = 0; r < 8; ++r) {
                int row = tm + 16 * i + r + 8 * half;
                out[(size_t)row * N + c] = acc[i][j][r] + bs;
            }
        }
    }
}

// ---------------------------------------------------------------------------
// RMSNorm + RoPE + layout reorg.  One wave per (b,h,n) row of D=64.
// Lane l owns d=l and d=l+32 (exactly the RoPE rotation pair, so no shuffles
// are needed for the rotation). Produces:
//   qb,kb : bf16 [B,H,N,D]   (row-major in D -> A/B frags for Q@K^T)
//   vt    : bf16 [B,H,D,N]   (row-major in N -> B frags for P@V)
// ---------------------------------------------------------------------------
__global__ __launch_bounds__(256)
void normrope(const float* __restrict__ qkv, const float* __restrict__ qn_w,
              const float* __restrict__ kn_w, const float* __restrict__ cosT,
              const float* __restrict__ sinT, bf16* __restrict__ qb,
              bf16* __restrict__ kb, bf16* __restrict__ vt) {
    const int lane = threadIdx.x & 31;
    const int wave = threadIdx.x >> 5;
    long row = (long)blockIdx.x * 8 + wave;   // (b*16 + h)*2048 + n
    int  b   = (int)(row >> 15);
    int  rem = (int)(row & 32767);
    int  h   = rem >> 11;
    int  n   = rem & 2047;

    const int d1 = lane, d2 = lane + 32;
    size_t base = ((size_t)b * 2048 + n) * 3072 + (size_t)h * 64;

    float c1 = cosT[n * 64 + d1], s1 = sinT[n * 64 + d1];
    float c2 = cosT[n * 64 + d2], s2 = sinT[n * 64 + d2];

    // ---- Q ----
    {
        float x1 = qkv[base + d1], x2 = qkv[base + d2];
        float ss = x1 * x1 + x2 * x2;
#pragma unroll
        for (int m = 1; m < 32; m <<= 1) ss += __shfl_xor(ss, m, 32);
        float r = rsqrtf(ss * (1.0f / 64.0f) + 1e-6f);
        float x1n = x1 * r * qn_w[d1];
        float x2n = x2 * r * qn_w[d2];
        size_t o = row * 64;
        qb[o + d1] = (bf16)(x1n * c1 - x2n * s1);
        qb[o + d2] = (bf16)(x2n * c2 + x1n * s2);
    }
    // ---- K ----
    {
        float x1 = qkv[base + 1024 + d1], x2 = qkv[base + 1024 + d2];
        float ss = x1 * x1 + x2 * x2;
#pragma unroll
        for (int m = 1; m < 32; m <<= 1) ss += __shfl_xor(ss, m, 32);
        float r = rsqrtf(ss * (1.0f / 64.0f) + 1e-6f);
        float x1n = x1 * r * kn_w[d1];
        float x2n = x2 * r * kn_w[d2];
        size_t o = row * 64;
        kb[o + d1] = (bf16)(x1n * c1 - x2n * s1);
        kb[o + d2] = (bf16)(x2n * c2 + x1n * s2);
    }
    // ---- V (store transposed [b,h,d,n]) ----
    {
        size_t vb = ((size_t)(b * 16 + h)) * 64 * 2048;
        vt[vb + (size_t)d1 * 2048 + n] = (bf16)qkv[base + 2048 + d1];
        vt[vb + (size_t)d2 * 2048 + n] = (bf16)qkv[base + 2048 + d2];
    }
}

// ---------------------------------------------------------------------------
// Flash attention. One wave per (b,h, 16-query-row tile). Streams keys in
// blocks of 32 with online softmax; all 4 matmul stages on v_wmma bf16.
// Output written as bf16 [B,N,H,D] (= row-major [B*N, C]) for the proj GEMM.
// ---------------------------------------------------------------------------
__global__ __launch_bounds__(256)
void flash_attn(const bf16* __restrict__ qb, const bf16* __restrict__ kb,
                const bf16* __restrict__ vt, bf16* __restrict__ attn_out) {
    __shared__ bf16 Pstage[8][16 * 32];

    const int lane = threadIdx.x & 31;
    const int wave = threadIdx.x >> 5;
    const int col  = lane & 15;
    const int half = lane >> 4;

    int wg = blockIdx.x * 8 + wave;   // (b*16+h)*128 + qtile
    int bh = wg >> 7;
    int q0 = (wg & 127) * 16;

    const bf16* Q  = qb + (size_t)bh * 2048 * 64;
    const bf16* Kp = kb + (size_t)bh * 2048 * 64;
    const bf16* V  = vt + (size_t)bh * 64 * 2048;

    // Q fragments (A layout), loop invariant: K-dim = D = 64 -> 2 frags.
    AFrag qa[2];
#pragma unroll
    for (int t = 0; t < 2; ++t) {
        const bf16* p = Q + (size_t)(q0 + col) * 64 + 32 * t + 8 * half;
        qa[t].h[0] = *(const bf16x8*)p;
        qa[t].h[1] = *(const bf16x8*)(p + 16);
    }

    v8f o[4] = {};
    float m[8], lsum[8];
#pragma unroll
    for (int r = 0; r < 8; ++r) { m[r] = -1e30f; lsum[r] = 0.f; }

    for (int kb0 = 0; kb0 < 2048; kb0 += 32) {
        // S = (Q @ K^T) for a 16x32 tile of scores (two 16x16 WMMA tiles)
        v8f s[2] = {};
#pragma unroll
        for (int j = 0; j < 2; ++j) {
#pragma unroll
            for (int t = 0; t < 2; ++t) {
                AFrag kf;
                kf.h[0] = *(const bf16x8*)(Kp + (size_t)(kb0 + 16 * j + col) * 64 +
                                           32 * t + 8 * half);
                kf.h[1] = *(const bf16x8*)(Kp + (size_t)(kb0 + 16 * j + col) * 64 +
                                           32 * t + 8 * half + 16);
                s[j] = __builtin_amdgcn_wmma_f32_16x16x32_bf16(
                    false, qa[t].v, false, kf.v, (short)0, s[j], false, false);
            }
        }

        // online softmax update per accumulator row r (M = r + 8*half)
        float p0[8], p1[8];
#pragma unroll
        for (int r = 0; r < 8; ++r) {
            float v0 = fmaxf(s[0][r], s[1][r]) * 0.125f;
#pragma unroll
            for (int msk = 1; msk < 16; msk <<= 1)
                v0 = fmaxf(v0, __shfl_xor(v0, msk, 32));
            float mn   = fmaxf(m[r], v0);
            float corr = __expf(m[r] - mn);
            m[r] = mn;
            float e0 = __expf(s[0][r] * 0.125f - mn);
            float e1 = __expf(s[1][r] * 0.125f - mn);
            float rs = e0 + e1;
#pragma unroll
            for (int msk = 1; msk < 16; msk <<= 1) rs += __shfl_xor(rs, msk, 32);
            lsum[r] = lsum[r] * corr + rs;
#pragma unroll
            for (int t = 0; t < 4; ++t) o[t][r] *= corr;
            p0[r] = e0;
            p1[r] = e1;
        }

        // stage P (C/D layout) -> LDS -> reload in A layout (DS in-order/wave)
        bf16* lp = Pstage[wave];
#pragma unroll
        for (int r = 0; r < 8; ++r) {
            int row = r + 8 * half;
            lp[row * 32 + col]      = (bf16)p0[r];
            lp[row * 32 + 16 + col] = (bf16)p1[r];
        }
        asm volatile("" ::: "memory");
        AFrag pa;
        {
            const bf16* p = lp + (size_t)col * 32 + 8 * half;
            pa.h[0] = *(const bf16x8*)p;
            pa.h[1] = *(const bf16x8*)(p + 16);
        }

        // O += P @ V  (V transposed -> contiguous B-frag loads)
#pragma unroll
        for (int t = 0; t < 4; ++t) {
            AFrag vf;
            vf.h[0] = *(const bf16x8*)(V + (size_t)(16 * t + col) * 2048 + kb0 +
                                       8 * half);
            vf.h[1] = *(const bf16x8*)(V + (size_t)(16 * t + col) * 2048 + kb0 +
                                       8 * half + 16);
            o[t] = __builtin_amdgcn_wmma_f32_16x16x32_bf16(
                false, pa.v, false, vf.v, (short)0, o[t], false, false);
        }
    }

    // normalize and store as [B,N,H,D]
    int b = bh >> 4, h = bh & 15;
#pragma unroll
    for (int r = 0; r < 8; ++r) {
        float inv = 1.0f / lsum[r];
        int qrow  = q0 + r + 8 * half;
        size_t base = ((size_t)(b * 2048 + qrow) * 16 + h) * 64;
#pragma unroll
        for (int t = 0; t < 4; ++t)
            attn_out[base + 16 * t + col] = (bf16)(o[t][r] * inv);
    }
}

// ---------------------------------------------------------------------------
// launch
// ---------------------------------------------------------------------------
extern "C" void kernel_launch(void* const* d_in, const int* in_sizes, int n_in,
                              void* d_out, int out_size, void* d_ws, size_t ws_size,
                              hipStream_t stream) {
    (void)in_sizes; (void)n_in; (void)out_size; (void)ws_size;
    const float* x      = (const float*)d_in[0];   // [4,2048,1024]
    const float* qkv_w  = (const float*)d_in[1];   // [3072,1024]
    const float* qkv_b  = (const float*)d_in[2];   // [3072]
    const float* proj_w = (const float*)d_in[3];   // [1024,1024]
    const float* proj_b = (const float*)d_in[4];   // [1024]
    const float* qn_w   = (const float*)d_in[5];   // [64]
    const float* kn_w   = (const float*)d_in[6];   // [64]
    const float* rcos   = (const float*)d_in[7];   // [2048,64]
    const float* rsin   = (const float*)d_in[8];   // [2048,64]
    float* out = (float*)d_out;                    // [8192,1024]

    const size_t MB = 1ull << 20;
    char* ws = (char*)d_ws;
    bf16*  x_bf    = (bf16*)(ws);              // 16 MB (reused as q_bf)
    bf16*  qkvw_bf = (bf16*)(ws + 16 * MB);    //  6 MB
    bf16*  projw_bf= (bf16*)(ws + 22 * MB);    //  2 MB
    float* qkv     = (float*)(ws + 24 * MB);   // 96 MB (reused as attn_bf)
    bf16*  k_bf    = (bf16*)(ws + 120 * MB);   // 16 MB
    bf16*  v_t     = (bf16*)(ws + 136 * MB);   // 16 MB -> total 152 MB
    bf16*  q_bf    = x_bf;                     // x_bf dead after QKV GEMM
    bf16*  attn_bf = (bf16*)qkv;               // qkv dead after normrope

    // 1) fp32 -> bf16 conversions
    cvt_f32_bf16<<<(8388608 + 255) / 256, 256, 0, stream>>>(x, x_bf, 8388608L);
    cvt_f32_bf16<<<(3145728 + 255) / 256, 256, 0, stream>>>(qkv_w, qkvw_bf, 3145728L);
    cvt_f32_bf16<<<(1048576 + 255) / 256, 256, 0, stream>>>(proj_w, projw_bf, 1048576L);

    // 2) QKV projection: [8192,1024] @ [3072,1024]^T -> [8192,3072]
    //    tiles: (8192/64)*(3072/64) = 6144 waves -> 768 blocks of 8 waves
    gemm_bf16_wmma<<<768, 256, 0, stream>>>(x_bf, qkvw_bf, qkv_b, qkv,
                                            8192, 3072, 1024);

    // 3) RMSNorm + RoPE + reorg: 4*16*2048 = 131072 rows, 8 waves/block
    normrope<<<16384, 256, 0, stream>>>(qkv, qn_w, kn_w, rcos, rsin,
                                        q_bf, k_bf, v_t);

    // 4) flash attention: 64 (b,h) * 128 q-tiles = 8192 waves -> 1024 blocks
    flash_attn<<<1024, 256, 0, stream>>>(q_bf, k_bf, v_t, attn_bf);

    // 5) output projection: [8192,1024] @ [1024,1024]^T -> d_out
    //    tiles: 128*16 = 2048 waves -> 256 blocks
    gemm_bf16_wmma<<<256, 256, 0, stream>>>(attn_bf, projw_bf, proj_b, out,
                                            8192, 1024, 1024);
}
